// LIIFHead_4767413698948
// MI455X (gfx1250) — matrix operand order repaired and checked by reading, not compiled, not run on hardware
//
#include <hip/hip_runtime.h>
#include <hip/hip_bf16.h>

typedef _Float16 v8h  __attribute__((ext_vector_type(8)));
typedef _Float16 v16h __attribute__((ext_vector_type(16)));
typedef float    v8f  __attribute__((ext_vector_type(8)));

#define B_   4
#define C_   64
#define H_   64
#define W_   64
#define N_   16384
#define K1P  608      // 580 padded up to 19*32
#define HID  256
#define TM   64       // rows (query-shift pairs) per workgroup
#define XSTR 616      // LDS X-tile row stride in halves (608 + pad)
#define ASTR 264      // LDS activation row stride in halves (256 + pad)

#define SMEM_ACTA  (TM * XSTR * 2)              // 78848:  X tile [64][616] f16
#define SMEM_META  (SMEM_ACTA + TM * ASTR * 2)  // 112640: actA [64][264] f16
#define SMEM_TOTAL (SMEM_META + 2048)           // 114688 bytes total

__device__ __forceinline__ v16h cat8(v8h lo, v8h hi) {
    return __builtin_shufflevector(lo, hi, 0,1,2,3,4,5,6,7,8,9,10,11,12,13,14,15);
}

// One MLP layer: each wave computes a 64x(NT*16) tile = MT row-blocks x NT col-tiles.
// Per k-step: MT A-fragments from LDS (shared across col-tiles), NT B-fragments from
// L2-resident pre-transposed weights Wt[n][k], MT*NT WMMAs.  MT=4/NT=2 gives 8 WMMAs
// per 2 global B loads -> 4x less L2 weight traffic than a 16-row-per-wave mapping.
template<int MT, int NT>
__device__ __forceinline__ void mlp_layer(
    const _Float16* __restrict__ lin, int istride, int ksteps,
    const _Float16* __restrict__ wt,  int wstride,
    const float*    __restrict__ bias,
    _Float16*       __restrict__ lout, int ostride,
    int wave, int lane)
{
    const int n0  = wave * (NT * 16);
    const int lhi = lane >> 4;
    const int llo = lane & 15;

    v8f acc[MT][NT];
#pragma unroll
    for (int r = 0; r < MT; ++r)
#pragma unroll
        for (int t = 0; t < NT; ++t)
            acc[r][t] = (v8f){0.f,0.f,0.f,0.f,0.f,0.f,0.f,0.f};

    for (int kk = 0; kk < ksteps; ++kk) {
        // A 16x32: lane half lhi holds K in [Kb,Kb+8) U [Kb+16,Kb+24), Kb = lhi*8
        v16h a[MT];
#pragma unroll
        for (int r = 0; r < MT; ++r) {
            const _Float16* ap = lin + (r * 16 + llo) * istride + kk * 32 + lhi * 8;
            a[r] = cat8(*(const v8h*)ap, *(const v8h*)(ap + 16));
        }
        // B 32x16: lane = column n, lane half lhi holds K in [Kb,Kb+8) U [Kb+16,Kb+24), Kb = lhi*8
        v16h bf[NT];
#pragma unroll
        for (int t = 0; t < NT; ++t) {
            const _Float16* bp = wt + (n0 + t * 16 + llo) * wstride + kk * 32 + lhi * 8;
            bf[t] = cat8(*(const v8h*)bp, *(const v8h*)(bp + 16));
        }
#pragma unroll
        for (int r = 0; r < MT; ++r)
#pragma unroll
            for (int t = 0; t < NT; ++t)
                acc[r][t] = __builtin_amdgcn_wmma_f32_16x16x32_f16(
                                false, a[r], false, bf[t], (short)0,
                                acc[r][t], false, false);
    }
    // C/D 16x16 f32: VGPR v, lanes 0-15 -> M=v, lanes 16-31 -> M=v+8; N = lane%16
#pragma unroll
    for (int t = 0; t < NT; ++t) {
        const int n = n0 + t * 16 + llo;
        const float bv = bias[n];
#pragma unroll
        for (int r = 0; r < MT; ++r)
#pragma unroll
            for (int v = 0; v < 8; ++v) {
                float x = acc[r][t][v] + bv;
                x = x > 0.f ? x : 0.f;
                lout[(r * 16 + v + lhi * 8) * ostride + n] = (_Float16)x;
            }
    }
}

// Transpose + f16-convert + zero-pad weights: dst[n*Kpad+k] = src[k*Ncols+n]
__global__ void wt_prep(const float* __restrict__ src, _Float16* __restrict__ dst,
                        int Krows, int Ncols, int Kpad, int Npad)
{
    int idx = blockIdx.x * 256 + threadIdx.x;
    if (idx >= Npad * Kpad) return;
    int n = idx / Kpad;
    int k = idx - n * Kpad;
    float v = 0.f;
    if (k < Krows && n < Ncols) v = src[k * Ncols + n];
    dst[idx] = (_Float16)v;
}

__global__ __launch_bounds__(256)
void liif_main(const float* __restrict__ feat,
               const float* __restrict__ coord,
               const float* __restrict__ cell,
               const float* __restrict__ b1, const float* __restrict__ b2,
               const float* __restrict__ b3, const float* __restrict__ b4,
               const float* __restrict__ b5,
               const _Float16* __restrict__ w1t, const _Float16* __restrict__ w2t,
               const _Float16* __restrict__ w3t, const _Float16* __restrict__ w4t,
               const _Float16* __restrict__ w5t,
               float* __restrict__ out)
{
    extern __shared__ char smem[];
    _Float16* Xb   = (_Float16*)smem;                 // [64][616] f16 input tile
    _Float16* actA = (_Float16*)(smem + SMEM_ACTA);   // [64][264] f16 activations
    _Float16* actB = Xb;                              // aliases X tile (dead after L1)
    float*    pred = (float*)actA;                    // [64][16] f32 (actA dead after L4)
    float*    m_rel  = (float*)(smem + SMEM_META);    // [64][4]
    float*    m_area = m_rel + TM * 4;                // [64]
    int*      m_iy   = (int*)(m_area + TM);
    int*      m_ix   = m_iy + TM;
    int*      m_bb   = m_ix + TM;

    const int tid = threadIdx.x;
    const int wg  = blockIdx.x;

    // ---- Phase 1: per-row metadata (nearest cell + relative coords) ----
    if (tid < TM) {
        int r = wg * TM + tid;
        int q = r >> 2;          // global query index b*N + n
        int s = r & 3;           // shift: s>>1 -> vx, s&1 -> vy (matches ref order)
        int b = q >> 14;         // q / N_
        float c0 = coord[q * 2 + 0], c1 = coord[q * 2 + 1];
        float e0 = cell[q * 2 + 0],  e1 = cell[q * 2 + 1];
        float vx = (s & 2) ? 1.f : -1.f;
        float vy = (s & 1) ? 1.f : -1.f;
        float cc0 = c0 + vx * (1.f / H_) + 1e-6f;
        float cc1 = c1 + vy * (1.f / W_) + 1e-6f;
        cc0 = fminf(fmaxf(cc0, -1.f + 1e-6f), 1.f - 1e-6f);
        cc1 = fminf(fmaxf(cc1, -1.f + 1e-6f), 1.f - 1e-6f);
        int iy = (int)rintf((cc0 + 1.f) * (H_ * 0.5f) - 0.5f);  // RNE = banker's
        int ix = (int)rintf((cc1 + 1.f) * (W_ * 0.5f) - 0.5f);
        iy = min(max(iy, 0), H_ - 1);
        ix = min(max(ix, 0), W_ - 1);
        float q0 = -1.f + (2.f * (float)iy + 1.f) / (float)H_;
        float q1 = -1.f + (2.f * (float)ix + 1.f) / (float)W_;
        float rel0 = (c0 - q0) * (float)H_;
        float rel1 = (c1 - q1) * (float)W_;
        m_rel[tid * 4 + 0] = rel0;
        m_rel[tid * 4 + 1] = rel1;
        m_rel[tid * 4 + 2] = e0 * (float)H_;
        m_rel[tid * 4 + 3] = e1 * (float)W_;
        m_area[tid] = fabsf(rel0 * rel1) + 1e-9f;
        m_iy[tid] = iy; m_ix[tid] = ix; m_bb[tid] = b;
    }
    __syncthreads();

    // ---- Phase 2: gather unfolded 3x3xC features into LDS X-tile (L2-bound) ----
    for (int e = tid; e < TM * 576; e += 256) {
        int i = e / 576;
        int f = e - i * 576;          // f = c*9 + k (channel-major, like unfold)
        int c = f / 9;
        int k = f - c * 9;
        int ki = k / 3;
        int kj = k - ki * 3;
        int y = m_iy[i] - 1 + ki;
        int x = m_ix[i] - 1 + kj;
        float v = 0.f;
        if ((unsigned)y < (unsigned)H_ && (unsigned)x < (unsigned)W_)
            v = feat[((m_bb[i] * C_ + c) * H_ + y) * W_ + x];
        Xb[i * XSTR + f] = (_Float16)v;
    }
    if (tid < TM) {
        Xb[tid * XSTR + 576] = (_Float16)m_rel[tid * 4 + 0];
        Xb[tid * XSTR + 577] = (_Float16)m_rel[tid * 4 + 1];
        Xb[tid * XSTR + 578] = (_Float16)m_rel[tid * 4 + 2];
        Xb[tid * XSTR + 579] = (_Float16)m_rel[tid * 4 + 3];
        for (int f = 580; f < XSTR; ++f) Xb[tid * XSTR + f] = (_Float16)0.f;
    }
    __syncthreads();

    const int wave = tid >> 5;
    const int lane = tid & 31;

    // ---- Phases 3-6: fused MLP layers via WMMA, LDS ping-pong ----
    mlp_layer<4, 2>(Xb,   XSTR, K1P / 32, w1t, K1P, b1, actA, ASTR, wave, lane);
    __syncthreads();
    mlp_layer<4, 2>(actA, ASTR, HID / 32, w2t, HID, b2, actB, ASTR, wave, lane);
    __syncthreads();
    mlp_layer<4, 2>(actB, ASTR, HID / 32, w3t, HID, b3, actA, ASTR, wave, lane);
    __syncthreads();
    mlp_layer<4, 2>(actA, ASTR, HID / 32, w4t, HID, b4, actB, ASTR, wave, lane);
    __syncthreads();

    // ---- Phase 7: final 256 -> 16 (3 valid) layer, f32 output to LDS ----
    if (wave < 4) {
        const int m0  = wave * 16;
        const int lhi = lane >> 4;
        const int llo = lane & 15;
        v8f acc = (v8f){0.f,0.f,0.f,0.f,0.f,0.f,0.f,0.f};
        for (int kk = 0; kk < HID / 32; ++kk) {
            const _Float16* ap = actB + (m0 + llo) * ASTR + kk * 32 + lhi * 8;
            v16h a = cat8(*(const v8h*)ap, *(const v8h*)(ap + 16));
            const _Float16* bp = w5t + llo * HID + kk * 32 + lhi * 8;
            v16h bf = cat8(*(const v8h*)bp, *(const v8h*)(bp + 16));
            acc = __builtin_amdgcn_wmma_f32_16x16x32_f16(
                      false, a, false, bf, (short)0, acc, false, false);
        }
        float bv = (llo < 3) ? b5[llo] : 0.f;
#pragma unroll
        for (int v = 0; v < 8; ++v)
            pred[(m0 + v + lhi * 8) * 16 + llo] = acc[v] + bv;
    }
    __syncthreads();

    // ---- Phase 8: local ensemble (diagonal area swap) ----
    if (tid < (TM / 4) * 3) {
        int ql = tid / 3;
        int c  = tid - ql * 3;
        int rb = ql * 4;
        float a0 = m_area[rb + 0], a1 = m_area[rb + 1];
        float a2 = m_area[rb + 2], a3 = m_area[rb + 3];
        float tot = a0 + a1 + a2 + a3;
        float ret = pred[(rb + 0) * 16 + c] * a3
                  + pred[(rb + 1) * 16 + c] * a2
                  + pred[(rb + 2) * 16 + c] * a1
                  + pred[(rb + 3) * 16 + c] * a0;
        out[(wg * (TM / 4) + ql) * 3 + c] = ret / tot;
    }
}

extern "C" void kernel_launch(void* const* d_in, const int* in_sizes, int n_in,
                              void* d_out, int out_size, void* d_ws, size_t ws_size,
                              hipStream_t stream) {
    const float* feat  = (const float*)d_in[0];
    const float* coord = (const float*)d_in[1];
    const float* cell  = (const float*)d_in[2];
    const float* W1 = (const float*)d_in[3];   const float* b1 = (const float*)d_in[4];
    const float* W2 = (const float*)d_in[5];   const float* b2 = (const float*)d_in[6];
    const float* W3 = (const float*)d_in[7];   const float* b3 = (const float*)d_in[8];
    const float* W4 = (const float*)d_in[9];   const float* b4 = (const float*)d_in[10];
    const float* W5 = (const float*)d_in[11];  const float* b5 = (const float*)d_in[12];

    // Workspace: pre-transposed f16 weights (~713 KB total)
    _Float16* w1t = (_Float16*)d_ws;           // [256][608]
    _Float16* w2t = w1t + 256 * K1P;           // [256][256]
    _Float16* w3t = w2t + 256 * 256;
    _Float16* w4t = w3t + 256 * 256;
    _Float16* w5t = w4t + 256 * 256;           // [16][256]

    wt_prep<<<(256 * K1P + 255) / 256, 256, 0, stream>>>(W1, w1t, 580, 256, K1P, 256);
    wt_prep<<<(256 * 256 + 255) / 256, 256, 0, stream>>>(W2, w2t, 256, 256, 256, 256);
    wt_prep<<<(256 * 256 + 255) / 256, 256, 0, stream>>>(W3, w3t, 256, 256, 256, 256);
    wt_prep<<<(256 * 256 + 255) / 256, 256, 0, stream>>>(W4, w4t, 256, 256, 256, 256);
    wt_prep<<<(16 * 256 + 255) / 256, 256, 0, stream>>>(W5, w5t, 256, 3, 256, 16);

    // Dynamic LDS of 112 KB exceeds the 64 KB default cap -> raise the limit.
    (void)hipFuncSetAttribute((const void*)liif_main,
                              hipFuncAttributeMaxDynamicSharedMemorySize, SMEM_TOTAL);

    const int rows = B_ * N_ * 4;              // 262144 query-shift rows
    liif_main<<<rows / TM, 256, SMEM_TOTAL, stream>>>(
        feat, coord, cell, b1, b2, b3, b4, b5,
        w1t, w2t, w3t, w4t, w5t, (float*)d_out);
}
